// Qwen3MoeTopKRouter_32049045963286
// MI455X (gfx1250) — matrix operand order repaired and checked by reading, not compiled, not run on hardware
//
#include <hip/hip_runtime.h>
#include <math.h>

typedef __attribute__((ext_vector_type(8)))  float  v8f;
typedef __attribute__((ext_vector_type(8)))  __bf16 v8bf;
typedef __attribute__((ext_vector_type(16))) __bf16 v16bf;

#define TOKENS      16384
#define HIDDEN      2048
#define NUM_EXPERTS 128
#define TOP_K       8

#define BM      64                  // tokens per block (4 M-tiles)
#define KC      256                 // K-chunk staged in LDS
#define APAD    24                  // pad -> row stride 280 bf16 = 140 dwords (== 12 mod 64)
#define ASTRIDE (KC + APAD)

// ---------------------------------------------------------------------------
// Split W (fp32) into bf16 hi + bf16 lo residual: w == wh + wl (to ~16 mantissa bits)
// ---------------------------------------------------------------------------
__global__ __launch_bounds__(256) void convert_w_kernel(
    const float* __restrict__ w, __bf16* __restrict__ wh, __bf16* __restrict__ wl)
{
  int i = blockIdx.x * 256 + threadIdx.x;          // NUM_EXPERTS*HIDDEN = 262144
  float f = w[i];
  __bf16 h = (__bf16)f;
  wh[i] = h;
  wl[i] = (__bf16)(f - (float)h);
}

// ---------------------------------------------------------------------------
// Router GEMM via 3-term bf16 split products:
//   logits ~= xh*wh + xh*wl + xl*wh   (xl*wl term ~2^-16 relative, dropped)
// Grid: TOKENS/BM blocks; 8 waves; wave w -> expert tile [16w,16w+16), loops
// 4 M-tiles. X converted to bf16 hi/lo in-register and staged in padded LDS.
// ---------------------------------------------------------------------------
__global__ __launch_bounds__(256) void router_gemm_bf16x3(
    const float* __restrict__ x, const __bf16* __restrict__ wh,
    const __bf16* __restrict__ wl, float* __restrict__ logits)
{
  __shared__ __bf16 Xh[BM * ASTRIDE];   // 35 KB
  __shared__ __bf16 Xl[BM * ASTRIDE];   // 35 KB

  const int tid  = threadIdx.x;
  const int wave = tid >> 5;
  const int lane = tid & 31;
  const int t0   = blockIdx.x * BM;
  const int e0   = wave * 16;
  const int n    = lane & 15;     // A: M row; B: N column
  const int kh   = lane >> 4;     // lane-half selects K sub-ranges

  const __bf16* whrow = wh + (size_t)(e0 + n) * HIDDEN;
  const __bf16* wlrow = wl + (size_t)(e0 + n) * HIDDEN;

  v8f c0 = {}, c1 = {}, c2 = {}, c3 = {};

  // cooperative load mapping: 4 threads per row, 8 octets (8 floats) each
  const int lr = tid >> 2;        // row 0..63
  const int lo = tid & 3;         // octet group within row

  for (int kc = 0; kc < HIDDEN; kc += KC) {
    const float* srow = x + (size_t)(t0 + lr) * HIDDEN + kc;
    if (kc + KC < HIDDEN) __builtin_prefetch(srow + KC, 0, 0);   // global_prefetch_b8

#pragma unroll
    for (int oc = 0; oc < 8; ++oc) {
      const int K0 = (lo + oc * 4) * 8;          // 8 consecutive floats
      float4 f0 = *(const float4*)(srow + K0);
      float4 f1 = *(const float4*)(srow + K0 + 4);
      float f[8] = {f0.x, f0.y, f0.z, f0.w, f1.x, f1.y, f1.z, f1.w};
      v8bf h, l;
#pragma unroll
      for (int j = 0; j < 8; ++j) {
        __bf16 hb = (__bf16)f[j];
        h[j] = hb;
        l[j] = (__bf16)(f[j] - (float)hb);
      }
      *(v8bf*)(Xh + lr * ASTRIDE + K0) = h;      // 16B ds_store
      *(v8bf*)(Xl + lr * ASTRIDE + K0) = l;
    }
    __syncthreads();

#pragma unroll 2
    for (int ks = 0; ks < KC; ks += 32) {
      // B frag (32x16 bf16): lane-half kh holds K in [16kh, 16kh+16), contiguous 32B
      v16bf bh = *(const v16bf*)(whrow + kc + ks + 16 * kh);
      v16bf bl = *(const v16bf*)(wlrow + kc + ks + 16 * kh);

#pragma unroll
      for (int mt = 0; mt < 4; ++mt) {
        // A frag (16x32 bf16): lane-half kh holds K in [16kh, 16kh+16), contiguous
        const __bf16* ah0 = Xh + (mt * 16 + n) * ASTRIDE + ks + 16 * kh;
        const __bf16* al0 = Xl + (mt * 16 + n) * ASTRIDE + ks + 16 * kh;
        v8bf ahA = *(const v8bf*)(ah0);
        v8bf ahB = *(const v8bf*)(ah0 + 8);
        v8bf alA = *(const v8bf*)(al0);
        v8bf alB = *(const v8bf*)(al0 + 8);
        v16bf ah = __builtin_shufflevector(ahA, ahB, 0,1,2,3,4,5,6,7,8,9,10,11,12,13,14,15);
        v16bf al = __builtin_shufflevector(alA, alB, 0,1,2,3,4,5,6,7,8,9,10,11,12,13,14,15);

        v8f& c = (mt == 0) ? c0 : (mt == 1) ? c1 : (mt == 2) ? c2 : c3;
        c = __builtin_amdgcn_wmma_f32_16x16x32_bf16(false, ah, false, bh, (short)0, c, false, false);
        c = __builtin_amdgcn_wmma_f32_16x16x32_bf16(false, ah, false, bl, (short)0, c, false, false);
        c = __builtin_amdgcn_wmma_f32_16x16x32_bf16(false, al, false, bh, (short)0, c, false, false);
      }
    }
    __syncthreads();
  }

  // C/D layout: VGPR i, lanes 0-15 -> M=i, N=lane; lanes 16-31 -> M=i+8
#pragma unroll
  for (int mt = 0; mt < 4; ++mt) {
    const v8f& c = (mt == 0) ? c0 : (mt == 1) ? c1 : (mt == 2) ? c2 : c3;
#pragma unroll
    for (int i = 0; i < 8; ++i) {
      logits[(size_t)(t0 + mt * 16 + i + 8 * kh) * NUM_EXPERTS + e0 + n] = c[i];
    }
  }
}

// ---------------------------------------------------------------------------
// Top-8 router: one wave per token (softmax monotone => top-8 of logits;
// renormalized top probs == softmax over the 8 selected logits).
// ---------------------------------------------------------------------------
__global__ __launch_bounds__(256) void router_topk(
    const float* __restrict__ logits, float* __restrict__ top_vals,
    int* __restrict__ top_idx)
{
  const int tid  = threadIdx.x;
  const int wave = tid >> 5;
  const int lane = tid & 31;
  const int t    = blockIdx.x * 8 + wave;

  const float* row = logits + (size_t)t * NUM_EXPERTS;
  float v[4];
#pragma unroll
  for (int j = 0; j < 4; ++j) v[j] = row[lane * 4 + j];

  float bv[TOP_K];
  int   bi[TOP_K];

#pragma unroll
  for (int it = 0; it < TOP_K; ++it) {
    float lv = v[0]; int li = lane * 4;
#pragma unroll
    for (int j = 1; j < 4; ++j) {
      if (v[j] > lv) { lv = v[j]; li = lane * 4 + j; }
    }
#pragma unroll
    for (int off = 16; off > 0; off >>= 1) {
      float ov = __shfl_xor(lv, off, 32);
      int   oi = __shfl_xor(li, off, 32);
      if (ov > lv || (ov == lv && oi < li)) { lv = ov; li = oi; }
    }
    bv[it] = lv; bi[it] = li;
    if ((li >> 2) == lane) v[li & 3] = -INFINITY;
  }

  const float m = bv[0];
  float e[TOP_K];
  float s = 0.f;
#pragma unroll
  for (int it = 0; it < TOP_K; ++it) { e[it] = __expf(bv[it] - m); s += e[it]; }
  const float inv = 1.0f / s;

  if (lane == 0) {
#pragma unroll
    for (int it = 0; it < TOP_K; ++it) {
      top_vals[(size_t)t * TOP_K + it] = e[it] * inv;
      top_idx [(size_t)t * TOP_K + it] = bi[it];
    }
  }
}

// ---------------------------------------------------------------------------
extern "C" void kernel_launch(void* const* d_in, const int* in_sizes, int n_in,
                              void* d_out, int out_size, void* d_ws, size_t ws_size,
                              hipStream_t stream) {
  (void)in_sizes; (void)n_in; (void)out_size; (void)ws_size;

  const float* x = (const float*)d_in[0];   // [TOKENS, HIDDEN]
  const float* w = (const float*)d_in[1];   // [NUM_EXPERTS, HIDDEN]

  float* out      = (float*)d_out;
  float* logits   = out;                                        // [TOKENS, NUM_EXPERTS]
  float* top_vals = out + (size_t)TOKENS * NUM_EXPERTS;         // [TOKENS, TOP_K]
  int*   top_idx  = (int*)(top_vals + (size_t)TOKENS * TOP_K);  // [TOKENS, TOP_K]

  // workspace: bf16 hi/lo split of W (2 * 512 KB = 1 MB)
  __bf16* wh = (__bf16*)d_ws;
  __bf16* wl = wh + (size_t)NUM_EXPERTS * HIDDEN;

  convert_w_kernel<<<(NUM_EXPERTS * HIDDEN) / 256, 256, 0, stream>>>(w, wh, wl);
  router_gemm_bf16x3<<<TOKENS / BM, 256, 0, stream>>>(x, wh, wl, logits);
  router_topk<<<TOKENS / 8, 256, 0, stream>>>(logits, top_vals, top_idx);
}
